// EgoProximityAgentAttention_78288663872324
// MI455X (gfx1250) — matrix operand order repaired and checked by reading, not compiled, not run on hardware
//
#include <hip/hip_runtime.h>
#include <hip/hip_bf16.h>

// ---------------------------------------------------------------------------
// EgoProximityAgentAttention for MI455X (gfx1250, wave32, WMMA + async-LDS)
// B=16, N=1024, D=512, NH=8, HD=64, K in {4,5,6} (device-computed)
// ---------------------------------------------------------------------------

#define BB 16
#define NN 1024
#define DD 512
#define NHH 8
#define HDD 64
#define CMAX 7          // K_MAX + 1 candidates per batch
#define LDA 520         // padded LDS row stride (halfwords): 1040B -> bank+4/row

typedef __attribute__((ext_vector_type(16))) __bf16        v16bf;
typedef __attribute__((ext_vector_type(8)))  float         v8f;
typedef __attribute__((ext_vector_type(4)))  unsigned int  u32x4;

union BF16Frag { v16bf v; u32x4 q[2]; };

__device__ __forceinline__ unsigned short f2bf_rne(float f) {
    unsigned int u = __float_as_uint(f);
    u += 0x7fffu + ((u >> 16) & 1u);   // round-to-nearest-even
    return (unsigned short)(u >> 16);
}

// ---------------------------------------------------------------------------
// Kernel 1: device-side K selection + per-batch top-7 nearest candidates.
// dist rows are identical within a batch (self masked later), so the global
// top-(K_MAX+1)=7 per batch determines every row's neighbor set.
// ---------------------------------------------------------------------------
__global__ void k_setup(const float* __restrict__ egoD,
                        const float* __restrict__ egoS,
                        int* __restrict__ wsK,
                        int* __restrict__ cIdx,
                        float* __restrict__ cDist) {
    __shared__ float red[256];
    __shared__ float dens;
    const int t = threadIdx.x;
    float cnt = 0.f;
    for (int i = t; i < BB * NN; i += 256) cnt += (egoD[i] < 20.0f) ? 1.f : 0.f;
    red[t] = cnt;
    __syncthreads();
    for (int s = 128; s > 0; s >>= 1) {
        if (t < s) red[t] += red[t + s];
        __syncthreads();
    }
    if (t == 0) {
        dens = red[0] * (1.0f / (BB * NN));
        float ss = 0.f;
        for (int b = 0; b < BB; ++b) ss += egoS[b];
        float sm = ss * (1.0f / BB);
        int K = 4;
        if (sm > 15.0f) K++;
        if (dens > 0.5f) K++;
        if (K > 6) K = 6;
        *wsK = K;
    }
    // top-7 smallest per batch, stable tie-break on lower index (lax.top_k)
    if (t < BB) {
        const float* dr = egoD + t * NN;
        int chosen[CMAX];
        for (int c = 0; c < CMAX; ++c) {
            float best = 3.4e38f;
            int bi = 0;
            for (int j = 0; j < NN; ++j) {
                bool used = false;
                for (int u = 0; u < c; ++u)
                    if (chosen[u] == j) used = true;
                if (used) continue;
                float v = dr[j];
                if (v < best) { best = v; bi = j; }
            }
            chosen[c] = bi;
            cIdx[t * 8 + c]  = bi;
            cDist[t * 8 + c] = best;
        }
    }
}

// ---------------------------------------------------------------------------
// Kernel 2: A (f32) -> bf16 staging (RNE)
// ---------------------------------------------------------------------------
__global__ void k_cvtA(const float* __restrict__ A, unsigned short* __restrict__ Abf) {
    const int i = blockIdx.x * 256 + threadIdx.x;
    if (i < BB * NN * DD) Abf[i] = f2bf_rne(A[i]);
}

// ---------------------------------------------------------------------------
// Kernel 3: Wq/Weq (f32, K-major) -> transposed bf16 Wt[n][k] so WMMA
// B-fragments become two contiguous 16B per-lane loads.
// ---------------------------------------------------------------------------
__global__ void k_cvtW(const float* __restrict__ Wq, const float* __restrict__ Weq,
                       unsigned short* __restrict__ Wqt, unsigned short* __restrict__ Weqt) {
    const int i = blockIdx.x * 256 + threadIdx.x;   // 0 .. 2*512*512-1
    const int mat = i >> 18;
    const int r = i & (DD * DD - 1);
    const int n = r >> 9;
    const int k = r & (DD - 1);
    const float* W = mat ? Weq : Wq;
    unsigned short* Wt = mat ? Weqt : Wqt;
    Wt[n * DD + k] = f2bf_rne(W[k * DD + n]);
}

// ---------------------------------------------------------------------------
// Kernel 4: K/V projections only for the <=7 candidate rows per batch.
// 112 rows x 512 x 512 x 2 -- negligible, plain VALU with LDS-staged row.
// ---------------------------------------------------------------------------
__global__ void k_candkv(const float* __restrict__ A, const int* __restrict__ cIdx,
                         const float* __restrict__ Wk, const float* __restrict__ bk,
                         const float* __restrict__ Wv, const float* __restrict__ bv,
                         float* __restrict__ Kc, float* __restrict__ Vc) {
    const int blk = blockIdx.x;         // b*7 + c
    const int b = blk / CMAX;
    const int c = blk % CMAX;
    const int row = cIdx[b * 8 + c];
    __shared__ float a[DD];
    const int t = threadIdx.x;
    const float* src = A + ((size_t)(b * NN + row)) * DD;
    a[t] = src[t];
    a[t + 256] = src[t + 256];
    __syncthreads();
    for (int cc = 0; cc < 2; ++cc) {
        const int col = t + cc * 256;
        float ak = bk[col], av = bv[col];
        for (int k = 0; k < DD; ++k) {
            const float x = a[k];
            ak = fmaf(x, Wk[k * DD + col], ak);
            av = fmaf(x, Wv[k * DD + col], av);
        }
        Kc[(size_t)blk * DD + col] = ak;
        Vc[(size_t)blk * DD + col] = av;
    }
}

// ---------------------------------------------------------------------------
// Kernel 5 (fused): async-DMA A tile to LDS -> WMMA Q-GEMM (Wq & Weq,
// per-row ego mix) -> LDS Q tile -> neighbor attention with multiplicative
// distance-MLP bias -> residual LayerNorm.
// One workgroup (8 wave32) per 16-row block; 1024 blocks.
// ---------------------------------------------------------------------------
__global__ __launch_bounds__(256) void k_fused(
    const unsigned short* __restrict__ Abf,
    const unsigned short* __restrict__ Wqt,
    const unsigned short* __restrict__ Weqt,
    const float* __restrict__ A,
    const float* __restrict__ egoD,
    const float* __restrict__ egoM,
    const float* __restrict__ bq,  const float* __restrict__ beq,
    const float* __restrict__ Wd1, const float* __restrict__ bd1,
    const float* __restrict__ Wd2, const float* __restrict__ bd2,
    const float* __restrict__ lng, const float* __restrict__ lnb,
    const int* __restrict__ pK,
    const int* __restrict__ cIdx, const float* __restrict__ cDist,
    const float* __restrict__ Kc, const float* __restrict__ Vc,
    float* __restrict__ out) {

    __shared__ float ldsQ[16][DD];                       // 32 KB
    __shared__ __align__(16) unsigned short ldsA[16 * LDA]; // 16.6 KB, padded
    __shared__ float red1[16][16], red2[16][16];         // 2 KB

    const int tid  = threadIdx.x;
    const int lane = tid & 31;
    const int wv   = tid >> 5;              // 0..7
    const int rowBase = blockIdx.x * 16;    // flat row (b*N + n)
    const int nlane = lane & 15;
    const int hi    = lane >> 4;            // 0 or 1

    // ---------------- Phase 0: async-DMA cols 0-255 of the 16x512 bf16 A tile into LDS ---
    // 512 chunks of 16B; each of the 256 threads issues 2 async copies.
    // EXEC is all-ones; ASYNCcnt tracks completion.
    for (int c = tid; c < 512; c += 256) {
        const int row = c >> 5;
        const int col = (c & 31) * 8;                       // halfword elems
        const unsigned short* g = Abf + (size_t)(rowBase + row) * DD + col;
        const unsigned int loff =
            (unsigned int)(size_t)&ldsA[row * LDA + col];   // addr[31:0] = LDS byte addr
        asm volatile("global_load_async_to_lds_b128 %0, %1, off"
                     :: "v"(loff), "v"(g)
                     : "memory");
    }
    asm volatile("s_wait_asynccnt 0" ::: "memory");
    __syncthreads();

    // ---------------- Phase 1: Q = mix(A@Wq, A@Weq) via bf16 WMMA ----------
    // A fragment from LDS: lane -> row M = lane&15; low lanes K octets
    // {0-7,16-23}, high lanes {8-15,24-31} (ISA 16-bit 16x32 A layout).
    // Padded stride LDA*2B = 260 dwords -> banks advance 4/row: conflict-free.
    const unsigned short* aLds = ldsA + nlane * LDA + (hi ? 8 : 0);
    const int kbase = hi ? 8 : 0;           // B layout: K octets match A (table order)

    v8f accq[4], acce[4];
    const v8f vzero = {0.f, 0.f, 0.f, 0.f, 0.f, 0.f, 0.f, 0.f};
#pragma unroll
    for (int t = 0; t < 4; ++t) { accq[t] = vzero; acce[t] = vzero; }

    for (int k0 = 0; k0 < DD; k0 += 32) {
        BF16Frag aF;
        aF.q[0] = *(const u32x4*)(aLds + k0);          // ds_load_b128
        aF.q[1] = *(const u32x4*)(aLds + k0 + 16);
#pragma unroll
        for (int t = 0; t < 4; ++t) {
            const int nc = (wv * 4 + t) * 16 + nlane;
            const unsigned short* pq = Wqt  + (size_t)nc * DD + k0 + kbase;
            const unsigned short* pe = Weqt + (size_t)nc * DD + k0 + kbase;
            BF16Frag bF, eF;
            bF.q[0] = *(const u32x4*)(pq);
            bF.q[1] = *(const u32x4*)(pq + 16);
            eF.q[0] = *(const u32x4*)(pe);
            eF.q[1] = *(const u32x4*)(pe + 16);
            accq[t] = __builtin_amdgcn_wmma_f32_16x16x32_bf16(
                false, aF.v, false, bF.v, (short)0, accq[t], false, false);
            acce[t] = __builtin_amdgcn_wmma_f32_16x16x32_bf16(
                false, aF.v, false, eF.v, (short)0, acce[t], false, false);
        }
    }

    // Epilogue: C/D layout c[r] = D[r + 8*hi][lane&15]; per-row ego mix + bias
#pragma unroll
    for (int t = 0; t < 4; ++t) {
        const int nc = (wv * 4 + t) * 16 + nlane;
        const float bqv = bq[nc], bev = beq[nc];
#pragma unroll
        for (int r = 0; r < 8; ++r) {
            const int mloc = r + hi * 8;
            const float e = egoM[rowBase + mloc];
            ldsQ[mloc][nc] = (1.f - e) * (accq[t][r] + bqv) + e * (acce[t][r] + bev);
        }
    }
    __syncthreads();

    // ---------------- Phase 2: neighbor attention (K<=6) -------------------
    const int sK = *pK;
    const int b = rowBase >> 10;        // / N
    const int nBase = rowBase & (NN - 1);

    if (tid < 128) {
        const int r = tid >> 3;         // local row
        const int h = tid & 7;          // head
        const int n = nBase + r;
        int slots[CMAX];
        int cnt = 0;
        for (int c = 0; c < CMAX && cnt < sK; ++c)
            if (cIdx[b * 8 + c] != n) slots[cnt++] = c;

        const float qd = egoD[b * NN + n];
        float sc[CMAX];
        for (int k = 0; k < cnt; ++k) {
            const int slot = slots[k];
            const float* kc = Kc + (size_t)(b * CMAX + slot) * DD + h * HDD;
            float s = 0.f;
            for (int d = 0; d < HDD; ++d) s = fmaf(ldsQ[r][h * HDD + d], kc[d], s);
            s *= 0.125f;   // 1/sqrt(64)
            // multiplicative distance bias: relu([qd,kd]@Wd1+bd1)@Wd2+bd2
            const float kd = cDist[b * 8 + slot];
            float bacc = bd2[h];
            for (int j = 0; j < 128; ++j) {
                float hh = fmaf(qd, Wd1[j], fmaf(kd, Wd1[128 + j], bd1[j]));
                hh = hh > 0.f ? hh : 0.f;
                bacc = fmaf(hh, Wd2[j * 8 + h], bacc);
            }
            sc[k] = s * bacc;
        }
        // softmax over k
        float mx = -3.4e38f;
        for (int k = 0; k < cnt; ++k) mx = fmaxf(mx, sc[k]);
        float sum = 0.f;
        for (int k = 0; k < cnt; ++k) { sc[k] = __expf(sc[k] - mx); sum += sc[k]; }
        const float inv = 1.f / sum;
        // attn @ V -- each thread reads/writes only its own ldsQ slice
        for (int d = 0; d < HDD; ++d) {
            float acc = 0.f;
            for (int k = 0; k < cnt; ++k)
                acc = fmaf(sc[k] * inv,
                           Vc[(size_t)(b * CMAX + slots[k]) * DD + h * HDD + d], acc);
            ldsQ[r][h * HDD + d] = acc;
        }
    }
    __syncthreads();

    // ---------------- Phase 3: residual + LayerNorm ------------------------
    const int r2 = tid >> 4;
    const int c2 = tid & 15;
    const int m = rowBase + r2;
    const float* aR = A + (size_t)m * DD;
    float s1 = 0.f, s2 = 0.f;
    for (int tt = 0; tt < 32; ++tt) {
        const int j = c2 + tt * 16;
        const float x = aR[j] + ldsQ[r2][j];
        s1 += x;
        s2 = fmaf(x, x, s2);
    }
    red1[r2][c2] = s1;
    red2[r2][c2] = s2;
    __syncthreads();
    float S1 = 0.f, S2 = 0.f;
    for (int u = 0; u < 16; ++u) { S1 += red1[r2][u]; S2 += red2[r2][u]; }
    const float mu  = S1 * (1.f / DD);
    const float var = S2 * (1.f / DD) - mu * mu;
    const float rs  = rsqrtf(var + 1e-5f);
    for (int tt = 0; tt < 32; ++tt) {
        const int j = c2 + tt * 16;
        const float x = aR[j] + ldsQ[r2][j];
        out[(size_t)m * DD + j] = (x - mu) * rs * lng[j] + lnb[j];
    }
}

// ---------------------------------------------------------------------------
// Host-side launch
// ---------------------------------------------------------------------------
extern "C" void kernel_launch(void* const* d_in, const int* in_sizes, int n_in,
                              void* d_out, int out_size, void* d_ws, size_t ws_size,
                              hipStream_t stream) {
    const float* A    = (const float*)d_in[0];
    const float* egoD = (const float*)d_in[1];
    const float* egoM = (const float*)d_in[2];
    const float* egoS = (const float*)d_in[3];
    const float* Wq   = (const float*)d_in[4];
    const float* bq   = (const float*)d_in[5];
    const float* Wk   = (const float*)d_in[6];
    const float* bk   = (const float*)d_in[7];
    const float* Wv   = (const float*)d_in[8];
    const float* bv   = (const float*)d_in[9];
    const float* Weq  = (const float*)d_in[10];
    const float* beq  = (const float*)d_in[11];
    const float* Wd1  = (const float*)d_in[16];
    const float* bd1  = (const float*)d_in[17];
    const float* Wd2  = (const float*)d_in[18];
    const float* bd2  = (const float*)d_in[19];
    const float* lng  = (const float*)d_in[20];
    const float* lnb  = (const float*)d_in[21];
    float* outp = (float*)d_out;

    char* ws = (char*)d_ws;
    const size_t OFF_K     = 0;
    const size_t OFF_CIDX  = 64;
    const size_t OFF_CDIST = 1024;
    const size_t OFF_KC    = 4096;
    const size_t OFF_VC    = OFF_KC + (size_t)BB * CMAX * DD * 4;
    const size_t OFF_ABF   = 524288;                                   // 512 KB aligned
    const size_t OFF_WQT   = OFF_ABF + (size_t)BB * NN * DD * 2;
    const size_t OFF_WEQT  = OFF_WQT + (size_t)DD * DD * 2;

    int*            wsK   = (int*)(ws + OFF_K);
    int*            cIdx  = (int*)(ws + OFF_CIDX);
    float*          cDist = (float*)(ws + OFF_CDIST);
    float*          Kc    = (float*)(ws + OFF_KC);
    float*          Vc    = (float*)(ws + OFF_VC);
    unsigned short* Abf   = (unsigned short*)(ws + OFF_ABF);
    unsigned short* Wqt   = (unsigned short*)(ws + OFF_WQT);
    unsigned short* Weqt  = (unsigned short*)(ws + OFF_WEQT);

    k_setup<<<1, 256, 0, stream>>>(egoD, egoS, wsK, cIdx, cDist);
    k_cvtA<<<(BB * NN * DD + 255) / 256, 256, 0, stream>>>(A, Abf);
    k_cvtW<<<(2 * DD * DD + 255) / 256, 256, 0, stream>>>(Wq, Weq, Wqt, Weqt);
    k_candkv<<<BB * CMAX, 256, 0, stream>>>(A, cIdx, Wk, bk, Wv, bv, Kc, Vc);
    k_fused<<<(BB * NN) / 16, 256, 0, stream>>>(Abf, Wqt, Weqt, A, egoD, egoM,
                                                bq, beq, Wd1, bd1, Wd2, bd2,
                                                lng, lnb, wsK, cIdx, cDist,
                                                Kc, Vc, outp);
}
